// NonLocalAttention_50354196578524
// MI455X (gfx1250) — matrix operand order repaired and checked by reading, not compiled, not run on hardware
//
#include <hip/hip_runtime.h>
#include <hip/hip_bf16.h>

typedef __attribute__((ext_vector_type(16))) _Float16 v16h;
typedef __attribute__((ext_vector_type(8)))  _Float16 v8h;
typedef __attribute__((ext_vector_type(8)))  float    v8f;

#define CCH 256      // channels
#define NPIX 4096    // H*W
#define BATCH 8
#define CRED 32      // reduced channels

// DPP16 row_ror:n within each 16-lane row (pure VALU cross-lane, no LDS)
#define ROW_ROR(x, n) __builtin_amdgcn_update_dpp(0, (x), 0x120 + (n), 0xf, 0xf, true)

__device__ __forceinline__ float row16_max(float v) {
    v = fmaxf(v, __int_as_float(ROW_ROR(__float_as_int(v), 1)));
    v = fmaxf(v, __int_as_float(ROW_ROR(__float_as_int(v), 2)));
    v = fmaxf(v, __int_as_float(ROW_ROR(__float_as_int(v), 4)));
    v = fmaxf(v, __int_as_float(ROW_ROR(__float_as_int(v), 8)));
    return v;
}

// ---------------------------------------------------------------------------
// Kernel 1: 1x1-conv projections  ->  f16 operands laid out for WMMA.
//   qT [B, N, 32]  (pre-scaled by 1/sqrt(32))
//   kT [B, N, 32]
//   v  [B, C, N]
// grid: (N/128, 10, B); block 256.  o-tile 0=q, 1=k, 2..9 = v rows.
// ---------------------------------------------------------------------------
__global__ void __launch_bounds__(256) proj_kernel(
    const float* __restrict__ x,
    const float* __restrict__ Wq, const float* __restrict__ bq,
    const float* __restrict__ Wk, const float* __restrict__ bk,
    const float* __restrict__ Wv, const float* __restrict__ bv,
    _Float16* __restrict__ qT, _Float16* __restrict__ kT,
    _Float16* __restrict__ vbuf)
{
    const int b  = blockIdx.z;
    const int ot = blockIdx.y;            // 0..9
    const int n0 = blockIdx.x * 128;
    const int t  = threadIdx.x;           // 0..255

    const float* W; const float* bias; int mode; int obase;
    if (ot == 0)      { W = Wq;                       bias = bq;                 mode = 0; obase = 0; }
    else if (ot == 1) { W = Wk;                       bias = bk;                 mode = 1; obase = 0; }
    else              { W = Wv + (ot - 2) * 32 * CCH; bias = bv + (ot - 2) * 32; mode = 2; obase = (ot - 2) * 32; }

    __shared__ float xs[64][128];
    __shared__ float Ws[32][64];

    const int o   = t >> 3;               // 0..31
    const int nj0 = (t & 7) * 16;         // 0..112
    float acc[16];
#pragma unroll
    for (int j = 0; j < 16; ++j) acc[j] = 0.f;

    const float* xb = x + (size_t)b * CCH * NPIX;

    for (int c0 = 0; c0 < CCH; c0 += 64) {
        __syncthreads();
        for (int i = t; i < 64 * 128; i += 256) {
            int cc = i >> 7, nn = i & 127;
            xs[cc][nn] = xb[(size_t)(c0 + cc) * NPIX + n0 + nn];
        }
        for (int i = t; i < 32 * 64; i += 256) {
            int oo = i >> 6, cc = i & 63;
            Ws[oo][cc] = W[oo * CCH + c0 + cc];
        }
        __syncthreads();
#pragma unroll 4
        for (int cc = 0; cc < 64; ++cc) {
            float wv = Ws[o][cc];
#pragma unroll
            for (int j = 0; j < 16; ++j)
                acc[j] = __builtin_fmaf(wv, xs[cc][nj0 + j], acc[j]);
        }
    }

    const float bval = bias[o];
    if (mode == 2) {
        _Float16* dst = vbuf + ((size_t)b * CCH + obase + o) * NPIX + n0 + nj0;
#pragma unroll
        for (int j = 0; j < 16; ++j) dst[j] = (_Float16)(acc[j] + bval);
    } else {
        const float s = (mode == 0) ? 0.17677669529663687f : 1.0f; // 1/sqrt(32)
        _Float16* dst = (mode == 0 ? qT : kT) + ((size_t)b * NPIX + n0 + nj0) * CRED + o;
#pragma unroll
        for (int j = 0; j < 16; ++j) dst[(size_t)j * CRED] = (_Float16)((acc[j] + bval) * s);
    }
}

// ---------------------------------------------------------------------------
// Kernel 2: fused flash attention + residual.
// A PAIR of wave32s shares one 16-row query tile; each wave owns 128 channels
// (8 f32 WMMA acc tiles) plus a "ones" tile for the softmax denominator.
// All 8 v-fragments for the step are loaded BEFORE the softmax so their L2
// latency hides behind the VALU/LDS work.
// grid: (N/16/4, B); block 256 (8 waves = 4 query tiles x 2 channel halves).
// ---------------------------------------------------------------------------
__global__ void __launch_bounds__(256) attn_kernel(
    const float* __restrict__ x, const float* __restrict__ gamma,
    const _Float16* __restrict__ qT, const _Float16* __restrict__ kT,
    const _Float16* __restrict__ vbuf, float* __restrict__ out)
{
    const int lane = threadIdx.x & 31;
    const int wave = threadIdx.x >> 5;     // 0..7
    const int qw   = wave >> 1;            // query tile within block: 0..3
    const int ch   = wave & 1;             // channel half: 0|1
    const int b    = blockIdx.y;
    const int n0   = (blockIdx.x * 4 + qw) * 16;

    __shared__ __align__(32) _Float16 sP[8][16 * 32];  // per-wave P staging
    __shared__ __align__(16) float    sT[8][16 * 16];  // per-wave epilogue transpose

    const int half = lane >> 4;   // 0|1
    const int l15  = lane & 15;

    // ---- A-matrix fragment of q (16 rows n, K=32 channels) ----
    union { v16h v; v8h h[2]; } qA;
    {
        const int row = n0 + l15;
        const int kb  = half ? 16 : 0;
        const _Float16* qp = qT + ((size_t)b * NPIX + row) * CRED + kb;
        qA.h[0] = *(const v8h*)(qp);
        qA.h[1] = *(const v8h*)(qp + 8);
    }

    // all-ones B fragment: WMMA(P, ones) accumulates row sums of P
    v16h onesB;
#pragma unroll
    for (int i = 0; i < 16; ++i) onesB[i] = (_Float16)1.0f;

    v8f acc[8];
    const v8f vzero = {};
#pragma unroll
    for (int t = 0; t < 8; ++t) acc[t] = vzero;
    v8f accL = vzero;                 // softmax denominator per row
    float m_st[8];
#pragma unroll
    for (int r = 0; r < 8; ++r) m_st[r] = -1e30f;

    const _Float16* kTb = kT + (size_t)b * NPIX * CRED;
    const _Float16* vb  = vbuf + (size_t)b * CCH * NPIX;
    const int kboff = half ? 16 : 0;

    // prime the k B-fragment pipeline (keys m0 .. m0+31)
    v16h kB0 = *(const v16h*)(kTb + (size_t)(0 + l15) * CRED + kboff);
    v16h kB1 = *(const v16h*)(kTb + (size_t)(16 + l15) * CRED + kboff);

    for (int m0 = 0; m0 < NPIX; m0 += 32) {
        // prefetch next iteration's k fragments (wrapped so last iter stays in-bounds)
        const int mn = (m0 + 32) & (NPIX - 1);
        v16h kB0n = *(const v16h*)(kTb + (size_t)(mn + l15) * CRED + kboff);
        v16h kB1n = *(const v16h*)(kTb + (size_t)(mn + 16 + l15) * CRED + kboff);

        v8f S0 = __builtin_amdgcn_wmma_f32_16x16x32_f16(false, qA.v, false, kB0,
                                                        (short)0, vzero, false, false);
        v8f S1 = __builtin_amdgcn_wmma_f32_16x16x32_f16(false, qA.v, false, kB1,
                                                        (short)0, vzero, false, false);

        // ---- issue ALL v-fragment loads now; latency hides behind softmax ----
        const int koff = m0 + kboff;
        v16h vL[8];
#pragma unroll
        for (int u = 0; u < 8; ++u)
            vL[u] = *(const v16h*)(vb + (size_t)((ch * 8 + u) * 16 + l15) * NPIX + koff);

        // ---- row maxima via DPP; rescale only on (rare) new-max events ----
        float mx8[8];
        bool anynew = false;
#pragma unroll
        for (int r = 0; r < 8; ++r) {
            mx8[r] = row16_max(fmaxf(S0[r], S1[r]));
            anynew |= (mx8[r] > m_st[r]);
        }
        if (__any(anynew)) {
#pragma unroll
            for (int r = 0; r < 8; ++r) {
                float mnew  = fmaxf(m_st[r], mx8[r]);
                float alpha = __expf(m_st[r] - mnew);
                m_st[r] = mnew;
                accL[r] *= alpha;
#pragma unroll
                for (int t = 0; t < 8; ++t) acc[t][r] *= alpha;
            }
        }

        // ---- P = exp(S - m); stage to LDS (C-layout -> row-major 16x32 f16) ----
#pragma unroll
        for (int r = 0; r < 8; ++r) {
            float p0 = __expf(S0[r] - m_st[r]);
            float p1 = __expf(S1[r] - m_st[r]);
            const int prow = r + 8 * half;
            sP[wave][prow * 32 + l15]      = (_Float16)p0;
            sP[wave][prow * 32 + 16 + l15] = (_Float16)p1;
        }
        asm volatile("s_wait_dscnt 0" ::: "memory");

        // re-read P as an A-matrix fragment (16 rows n, K=32 keys)
        union { v16h v; v8h h[2]; } pA;
        {
            const int cb = half ? 16 : 0;
            const _Float16* pp = &sP[wave][l15 * 32 + cb];
            pA.h[0] = *(const v8h*)(pp);
            pA.h[1] = *(const v8h*)(pp + 8);
        }

        // denominator accumulation: rowsum(P) via WMMA with all-ones B
        accL = __builtin_amdgcn_wmma_f32_16x16x32_f16(false, pA.v, false, onesB,
                                                      (short)0, accL, false, false);

        // ---- out += P x V^T over this wave's 8 channel tiles ----
#pragma unroll
        for (int u = 0; u < 8; ++u) {
            acc[u] = __builtin_amdgcn_wmma_f32_16x16x32_f16(
                false, pA.v, false, vL[u], (short)0, acc[u], false, false);
        }

        kB0 = kB0n;
        kB1 = kB1n;
    }

    // ---- epilogue: normalize, transpose via LDS, residual add, store ----
    const float g = gamma[0];
    float rinv[8];
#pragma unroll
    for (int r = 0; r < 8; ++r) rinv[r] = 1.0f / accL[r];

#pragma unroll
    for (int t = 0; t < 8; ++t) {
#pragma unroll
        for (int r = 0; r < 8; ++r)
            sT[wave][(r + 8 * half) * 16 + l15] = acc[t][r] * rinv[r];
        asm volatile("s_wait_dscnt 0" ::: "memory");

        const int cg = (ch * 8 + t) * 16 + l15;
        const float* xp = x   + ((size_t)b * CCH + cg) * NPIX + n0 + 8 * half;
        float*       op = out + ((size_t)b * CCH + cg) * NPIX + n0 + 8 * half;
#pragma unroll
        for (int j = 0; j < 8; ++j) {
            float val = sT[wave][(8 * half + j) * 16 + l15];
            op[j] = xp[j] + g * val;
        }
        asm volatile("s_wait_dscnt 0" ::: "memory");
    }
}

extern "C" void kernel_launch(void* const* d_in, const int* in_sizes, int n_in,
                              void* d_out, int out_size, void* d_ws, size_t ws_size,
                              hipStream_t stream) {
    const float* x     = (const float*)d_in[0];
    const float* Wq    = (const float*)d_in[1];
    const float* bq    = (const float*)d_in[2];
    const float* Wk    = (const float*)d_in[3];
    const float* bk    = (const float*)d_in[4];
    const float* Wv    = (const float*)d_in[5];
    const float* bv    = (const float*)d_in[6];
    const float* gamma = (const float*)d_in[7];
    float* out = (float*)d_out;

    _Float16* qT = (_Float16*)d_ws;                       // 2 MB
    _Float16* kT = qT + (size_t)BATCH * NPIX * CRED;      // 2 MB
    _Float16* vb = kT + (size_t)BATCH * NPIX * CRED;      // 16 MB

    dim3 g1(NPIX / 128, 10, BATCH);
    proj_kernel<<<g1, 256, 0, stream>>>(x, Wq, bq, Wk, bk, Wv, bv, qT, kT, vb);

    dim3 g2(NPIX / 16 / 4, BATCH, 1);
    attn_kernel<<<g2, 256, 0, stream>>>(x, gamma, qT, kT, vb, out);
}
